// MultiHeadAttention_62414464745765
// MI455X (gfx1250) — matrix operand order repaired and checked by reading, not compiled, not run on hardware
//
#include <hip/hip_runtime.h>
#include <hip/hip_bf16.h>

// ---- problem constants (match reference) ----
constexpr int kB  = 2;
constexpr int kS  = 2048;
constexpr int kD  = 1024;
constexpr int kH  = 16;
constexpr int kDK = 64;
constexpr float kNEG = -1000000000.0f;

typedef __attribute__((ext_vector_type(16))) __bf16 v16bf;
typedef __attribute__((ext_vector_type(8)))  __bf16 v8bf;
typedef __attribute__((ext_vector_type(8)))  float  v8f;
typedef int v4i __attribute__((__vector_size__(16)));   // matches builtin param type

#define AS_GLOBAL __attribute__((address_space(1)))
#define AS_LOCAL  __attribute__((address_space(3)))

#if defined(__AMDGCN__) && __has_builtin(__builtin_amdgcn_global_load_async_to_lds_b128) && \
    __has_builtin(__builtin_amdgcn_s_wait_asynccnt)
#define HAVE_ASYNC_LDS 1
#else
#define HAVE_ASYNC_LDS 0
#endif

#if HAVE_ASYNC_LDS
// async 16B global->LDS copy (ASYNCcnt tracked); no VGPR data round-trip
__device__ __forceinline__ void async_cp16(const __bf16* gsrc, __bf16* ldst) {
    __builtin_amdgcn_global_load_async_to_lds_b128(
        (AS_GLOBAL v4i*)gsrc, (AS_LOCAL v4i*)ldst, 0, 0);
}
__device__ __forceinline__ void async_wait0() {
    __builtin_amdgcn_s_wait_asynccnt(0);
}
#endif

// D = A(16x32 bf16) * B(32x16 bf16) + C(16x16 f32)
__device__ __forceinline__ v8f wmma_bf16(v16bf a, v16bf b, v8f c) {
    return __builtin_amdgcn_wmma_f32_16x16x32_bf16(
        /*neg_a=*/false, a, /*neg_b=*/false, b,
        /*c_mod=*/(short)0, c, /*reuse_a=*/false, /*reuse_b=*/false);
}

// Load a 16-element bf16 fragment as two contiguous 8-element (16B) runs.
__device__ __forceinline__ v16bf ld16(const __bf16* p0, const __bf16* p1) {
    v8bf lo = *(const v8bf*)p0;
    v8bf hi = *(const v8bf*)p1;
    v16bf r;
#pragma unroll
    for (int i = 0; i < 8; ++i) { r[i] = lo[i]; r[8 + i] = hi[i]; }
    return r;
}

// Row reductions across the 16-lane half (C-layout rows live in lane halves).
__device__ __forceinline__ float rowmax16(float v) {
#pragma unroll
    for (int m = 1; m < 16; m <<= 1) v = fmaxf(v, __shfl_xor(v, m, 32));
    return v;
}
__device__ __forceinline__ float rowsum16(float v) {
#pragma unroll
    for (int m = 1; m < 16; m <<= 1) v += __shfl_xor(v, m, 32);
    return v;
}

// ---------------------------------------------------------------------------
// Kernel 1: Q/K projection.  [B*S, D] @ [D, D] + bias, fp32 in -> bf16 out,
// scattered into [B, H, S, DK] layout.  blockIdx.z selects (Wq,bq)/(Wk,bk).
// Block: 256 threads = 8 waves; tile 128(M) x 64(N); K-step 32.
// (fp32->bf16 conversion happens in VGPRs, so this path stays synchronous.)
// ---------------------------------------------------------------------------
__global__ void mha_proj_kernel(const float* __restrict__ x,
                                const float* __restrict__ Wq, const float* __restrict__ bq,
                                const float* __restrict__ Wk, const float* __restrict__ bk,
                                __bf16* __restrict__ qout, __bf16* __restrict__ kout) {
    const int which = blockIdx.z;
    const float* __restrict__ W  = which ? Wk : Wq;
    const float* __restrict__ bv = which ? bk : bq;
    __bf16* __restrict__ out     = which ? kout : qout;

    const int n0      = blockIdx.x * 64;
    const int rowBase = blockIdx.y * 128;
    const int tid  = threadIdx.x;
    const int w    = tid >> 5;          // wave 0..7
    const int lane = tid & 31;
    const int g    = lane >> 4;         // lane half
    const int n16  = lane & 15;
    const int wr0  = w * 16;            // wave's row strip in the tile

    __shared__ __align__(16) __bf16 As[128][40];   // x tile, row-major (K stride 40)
    __shared__ __align__(16) __bf16 BTs[64][40];   // W tile, TRANSPOSED: [n][k]

    v8f acc[4];
#pragma unroll
    for (int c = 0; c < 4; ++c)
#pragma unroll
        for (int r = 0; r < 8; ++r) acc[c][r] = 0.0f;

    for (int kt = 0; kt < kD / 32; ++kt) {
        const int k0 = kt * 32;
        // stage A: 128x32 fp32 -> bf16 (float4 loads)
#pragma unroll
        for (int it = 0; it < 4; ++it) {
            int slot = tid + it * 256;          // 0..1023
            int i = slot >> 3, j4 = (slot & 7) * 4;
            float4 v = *(const float4*)&x[(size_t)(rowBase + i) * kD + k0 + j4];
            As[i][j4 + 0] = (__bf16)v.x; As[i][j4 + 1] = (__bf16)v.y;
            As[i][j4 + 2] = (__bf16)v.z; As[i][j4 + 3] = (__bf16)v.w;
        }
        // stage B transposed: rows k0..k0+31, cols n0..n0+63
#pragma unroll
        for (int it = 0; it < 2; ++it) {
            int slot = tid + it * 256;          // 0..511
            int kk = slot >> 4, j4 = (slot & 15) * 4;
            float4 v = *(const float4*)&W[(size_t)(k0 + kk) * kD + n0 + j4];
            BTs[j4 + 0][kk] = (__bf16)v.x; BTs[j4 + 1][kk] = (__bf16)v.y;
            BTs[j4 + 2][kk] = (__bf16)v.z; BTs[j4 + 3][kk] = (__bf16)v.w;
        }
        __syncthreads();

        if (kt + 1 < kD / 32) {  // prefetch next tiles (global_prefetch_b8)
            __builtin_prefetch(&x[(size_t)(rowBase + (tid >> 1)) * kD + k0 + 32 + (tid & 1) * 16], 0, 1);
            __builtin_prefetch(&W[(size_t)(k0 + 32 + (tid >> 3)) * kD + n0 + (tid & 7) * 8], 0, 1);
        }

        v16bf a = ld16(&As[wr0 + n16][g * 16], &As[wr0 + n16][g * 16 + 8]);
#pragma unroll
        for (int c = 0; c < 4; ++c) {
            v16bf b = ld16(&BTs[c * 16 + n16][g * 16], &BTs[c * 16 + n16][g * 16 + 8]);
            acc[c] = wmma_bf16(a, b, acc[c]);
        }
        __syncthreads();
    }

    // epilogue: + bias, scatter to [B, H, S, DK] as bf16
#pragma unroll
    for (int c = 0; c < 4; ++c) {
        const int colg = n0 + c * 16 + n16;
        const float bias = bv[colg];
        const int h = colg >> 6, dk = colg & 63;
#pragma unroll
        for (int r = 0; r < 8; ++r) {
            int rowg = rowBase + wr0 + r + 8 * g;
            int bidx = rowg / kS, s = rowg % kS;
            out[((size_t)(bidx * kH + h) * kS + s) * kDK + dk] = (__bf16)(acc[c][r] + bias);
        }
    }
}

// ---------------------------------------------------------------------------
// Kernel 2: flash attention with causal mask and online softmax. V == Q (bug
// replicated).  Block: 128 threads = 4 waves; each wave owns 16 q rows;
// k processed in blocks of 32.  Writes att (bf16) in [B, S, D] layout.
// Q/K tile staging uses the gfx1250 async global->LDS engine when available.
// ---------------------------------------------------------------------------
__global__ void mha_flash_kernel(const __bf16* __restrict__ q,
                                 const __bf16* __restrict__ k,
                                 __bf16* __restrict__ att) {
    const int tq   = blockIdx.x;        // q block of 64 rows
    const int bh   = blockIdx.y;
    const int bidx = bh / kH, h = bh % kH;
    const int tid  = threadIdx.x;
    const int w    = tid >> 5, lane = tid & 31;
    const int g    = lane >> 4, n16 = lane & 15;
    const int qbase = tq * 64 + w * 16;

    const __bf16* __restrict__ qh = q + (size_t)(bidx * kH + h) * kS * kDK;
    const __bf16* __restrict__ kh = k + (size_t)(bidx * kH + h) * kS * kDK;

    __shared__ __align__(16) __bf16 Qs[64][72];       // q tile, row-major
    __shared__ __align__(16) __bf16 Ks[32][72];       // k tile, row-major
    __shared__ __align__(16) __bf16 VTs[64][40];      // v tile TRANSPOSED [dk][krow]
    __shared__ __align__(16) __bf16 Ps[4][16][40];    // per-wave P scratch

    // stage Q tile 64x64 once
#if HAVE_ASYNC_LDS
#pragma unroll
    for (int it = 0; it < 4; ++it) {
        int slot = tid + it * 128;                    // 0..511
        int i = slot >> 3, j8 = (slot & 7) * 8;
        async_cp16(&qh[(size_t)(tq * 64 + i) * kDK + j8], &Qs[i][j8]);
    }
    async_wait0();
#else
#pragma unroll
    for (int it = 0; it < 4; ++it) {
        int slot = tid + it * 128;                    // 0..511
        int i = slot >> 3, j8 = (slot & 7) * 8;
        *(v8bf*)&Qs[i][j8] = *(const v8bf*)&qh[(size_t)(tq * 64 + i) * kDK + j8];
    }
#endif
    __syncthreads();

    // invariant Q A-fragments (dk chunks 0..31 and 32..63)
    const v16bf aq0 = ld16(&Qs[w * 16 + n16][0 + g * 16],  &Qs[w * 16 + n16][g * 16 + 8]);
    const v16bf aq1 = ld16(&Qs[w * 16 + n16][32 + g * 16], &Qs[w * 16 + n16][32 + g * 16 + 8]);

    v8f oacc[4];
    float mrow[8], lrow[8];
#pragma unroll
    for (int c = 0; c < 4; ++c)
#pragma unroll
        for (int r = 0; r < 8; ++r) oacc[c][r] = 0.0f;
#pragma unroll
    for (int r = 0; r < 8; ++r) { mrow[r] = -3.0e38f; lrow[r] = 0.0f; }

    const int kend = tq * 64 + 64;
    for (int k0 = 0; k0 < kend; k0 += 32) {
        // stage K tile (row-major; async engine) and V tile (= Q rows,
        // transposed in VGPRs, overlapped with the async K copy)
#pragma unroll
        for (int it = 0; it < 2; ++it) {
            int slot = tid + it * 128;                // 0..255
            int kk = slot >> 3, j8 = (slot & 7) * 8;
#if HAVE_ASYNC_LDS
            async_cp16(&kh[(size_t)(k0 + kk) * kDK + j8], &Ks[kk][j8]);
#else
            *(v8bf*)&Ks[kk][j8] = *(const v8bf*)&kh[(size_t)(k0 + kk) * kDK + j8];
#endif
            v8bf vq = *(const v8bf*)&qh[(size_t)(k0 + kk) * kDK + j8];
#pragma unroll
            for (int e = 0; e < 8; ++e) VTs[j8 + e][kk] = vq[e];
        }
#if HAVE_ASYNC_LDS
        async_wait0();
#endif
        __syncthreads();

        if (k0 <= qbase + 15) {                       // wave-uniform guard: EXEC stays all-ones
            // ---- scores: two 16x16 tiles, contraction dk=64 in two steps ----
            v8f st[2];
#pragma unroll
            for (int c = 0; c < 2; ++c) {
#pragma unroll
                for (int r = 0; r < 8; ++r) st[c][r] = 0.0f;
                v16bf b0 = ld16(&Ks[c * 16 + n16][g * 16],      &Ks[c * 16 + n16][g * 16 + 8]);
                st[c] = wmma_bf16(aq0, b0, st[c]);
                v16bf b1 = ld16(&Ks[c * 16 + n16][32 + g * 16], &Ks[c * 16 + n16][32 + g * 16 + 8]);
                st[c] = wmma_bf16(aq1, b1, st[c]);
            }
            // ---- scale + causal mask ----
#pragma unroll
            for (int c = 0; c < 2; ++c)
#pragma unroll
                for (int r = 0; r < 8; ++r) {
                    int qi = qbase + r + 8 * g;
                    int ki = k0 + c * 16 + n16;
                    float s = st[c][r] * 0.125f;      // 1/sqrt(64)
                    st[c][r] = (ki > qi) ? kNEG : s;
                }
            // ---- online softmax ----
            float mnew[8], alpha[8], psum[8];
#pragma unroll
            for (int r = 0; r < 8; ++r) {
                float t = rowmax16(fmaxf(st[0][r], st[1][r]));
                mnew[r]  = fmaxf(t, mrow[r]);
                alpha[r] = __expf(mrow[r] - mnew[r]);
                mrow[r]  = mnew[r];
                psum[r]  = 0.0f;
            }
#pragma unroll
            for (int c = 0; c < 2; ++c)
#pragma unroll
                for (int r = 0; r < 8; ++r) {
                    float p = __expf(st[c][r] - mnew[r]);
                    psum[r] += p;
                    Ps[w][r + 8 * g][c * 16 + n16] = (__bf16)p;   // re-swizzle via LDS
                }
#pragma unroll
            for (int r = 0; r < 8; ++r) lrow[r] = lrow[r] * alpha[r] + rowsum16(psum[r]);
#pragma unroll
            for (int c = 0; c < 4; ++c)
#pragma unroll
                for (int r = 0; r < 8; ++r) oacc[c][r] *= alpha[r];

            __asm__ volatile("" ::: "memory");        // keep P stores before A-frag loads
            // ---- O += P(16x32) @ V(32x64) ----
            v16bf ap = ld16(&Ps[w][n16][g * 16], &Ps[w][n16][g * 16 + 8]);
#pragma unroll
            for (int c = 0; c < 4; ++c) {
                v16bf bvv = ld16(&VTs[c * 16 + n16][g * 16], &VTs[c * 16 + n16][g * 16 + 8]);
                oacc[c] = wmma_bf16(ap, bvv, oacc[c]);
            }
        }
        __syncthreads();
    }

    // epilogue: normalize and write att in [B, S, D] layout (bf16)
#pragma unroll
    for (int r = 0; r < 8; ++r) {
        const float inv = 1.0f / lrow[r];
        const int s_idx = qbase + r + 8 * g;
#pragma unroll
        for (int c = 0; c < 4; ++c) {
            int d = h * kDK + c * 16 + n16;
            att[(size_t)(bidx * kS + s_idx) * kD + d] = (__bf16)(oacc[c][r] * inv);
        }
    }
}

// ---------------------------------------------------------------------------
// Kernel 3: output projection.  att(bf16) [B*S, D] @ W0 + b0 -> fp32 out.
// A tile is a pure bf16 copy -> async global->LDS engine when available.
// ---------------------------------------------------------------------------
__global__ void mha_outproj_kernel(const __bf16* __restrict__ att,
                                   const float* __restrict__ W0,
                                   const float* __restrict__ b0,
                                   float* __restrict__ out) {
    const int n0      = blockIdx.x * 64;
    const int rowBase = blockIdx.y * 128;
    const int tid  = threadIdx.x;
    const int w    = tid >> 5, lane = tid & 31;
    const int g    = lane >> 4, n16 = lane & 15;
    const int wr0  = w * 16;

    __shared__ __align__(16) __bf16 As[128][40];
    __shared__ __align__(16) __bf16 BTs[64][40];

    v8f acc[4];
#pragma unroll
    for (int c = 0; c < 4; ++c)
#pragma unroll
        for (int r = 0; r < 8; ++r) acc[c][r] = 0.0f;

    for (int kt = 0; kt < kD / 32; ++kt) {
        const int k0 = kt * 32;
        // stage A (already bf16): 128x32
#pragma unroll
        for (int it = 0; it < 2; ++it) {
            int slot = tid + it * 256;                // 0..511
            int i = slot >> 2, j8 = (slot & 3) * 8;
#if HAVE_ASYNC_LDS
            async_cp16(&att[(size_t)(rowBase + i) * kD + k0 + j8], &As[i][j8]);
#else
            *(v8bf*)&As[i][j8] = *(const v8bf*)&att[(size_t)(rowBase + i) * kD + k0 + j8];
#endif
        }
        // stage W0 transposed (fp32 -> bf16 in VGPRs, overlaps async A copy)
#pragma unroll
        for (int it = 0; it < 2; ++it) {
            int slot = tid + it * 256;
            int kk = slot >> 4, j4 = (slot & 15) * 4;
            float4 v = *(const float4*)&W0[(size_t)(k0 + kk) * kD + n0 + j4];
            BTs[j4 + 0][kk] = (__bf16)v.x; BTs[j4 + 1][kk] = (__bf16)v.y;
            BTs[j4 + 2][kk] = (__bf16)v.z; BTs[j4 + 3][kk] = (__bf16)v.w;
        }
#if HAVE_ASYNC_LDS
        async_wait0();
#endif
        __syncthreads();

        if (kt + 1 < kD / 32) {
            __builtin_prefetch(&att[(size_t)(rowBase + (tid >> 1)) * kD + k0 + 32 + (tid & 1) * 16], 0, 1);
            __builtin_prefetch(&W0[(size_t)(k0 + 32 + (tid >> 3)) * kD + n0 + (tid & 7) * 8], 0, 1);
        }

        v16bf a = ld16(&As[wr0 + n16][g * 16], &As[wr0 + n16][g * 16 + 8]);
#pragma unroll
        for (int c = 0; c < 4; ++c) {
            v16bf b = ld16(&BTs[c * 16 + n16][g * 16], &BTs[c * 16 + n16][g * 16 + 8]);
            acc[c] = wmma_bf16(a, b, acc[c]);
        }
        __syncthreads();
    }

#pragma unroll
    for (int c = 0; c < 4; ++c) {
        const int colg = n0 + c * 16 + n16;
        const float bias = b0[colg];
#pragma unroll
        for (int r = 0; r < 8; ++r) {
            int rowg = rowBase + wr0 + r + 8 * g;
            out[(size_t)rowg * kD + colg] = acc[c][r] + bias;
        }
    }
}

// ---------------------------------------------------------------------------
extern "C" void kernel_launch(void* const* d_in, const int* in_sizes, int n_in,
                              void* d_out, int out_size, void* d_ws, size_t ws_size,
                              hipStream_t stream) {
    (void)in_sizes; (void)n_in; (void)out_size; (void)ws_size;
    const float* x  = (const float*)d_in[0];
    const float* Wq = (const float*)d_in[1];
    const float* bq = (const float*)d_in[2];
    const float* Wk = (const float*)d_in[3];
    const float* bk = (const float*)d_in[4];
    const float* W0 = (const float*)d_in[5];
    const float* b0 = (const float*)d_in[6];
    float* out = (float*)d_out;

    // workspace: q, k in [B,H,S,DK]; att in [B,S,D]; all bf16 (24 MB total)
    __bf16* qws = (__bf16*)d_ws;
    __bf16* kws = qws + (size_t)kB * kS * kD;
    __bf16* aws = kws + (size_t)kB * kS * kD;

    mha_proj_kernel<<<dim3(kD / 64, (kB * kS) / 128, 2), 256, 0, stream>>>(
        x, Wq, bq, Wk, bk, qws, kws);
    mha_flash_kernel<<<dim3(kS / 64, kB * kH), 128, 0, stream>>>(qws, kws, aws);
    mha_outproj_kernel<<<dim3(kD / 64, (kB * kS) / 128), 256, 0, stream>>>(
        aws, W0, b0, out);
}
